// AutoregressiveArithmeticTransformer_51659866636729
// MI455X (gfx1250) — compile-verified, matrix operand order checked by reading, not run on hardware
//
#include <hip/hip_runtime.h>

// ---------------- constants (match reference) ----------------
constexpr int cV  = 16;
constexpr int cE  = 512;
constexpr int cNH = 8;
constexpr int cHS = 64;
constexpr int cFF = 2048;
constexpr int cNB = 6;
constexpr int cL  = 512;
constexpr int cB  = 16;
constexpr int cT  = 512;
constexpr int cM  = cB * cT;          // 8192 rows of the activation matrix

typedef __attribute__((ext_vector_type(16))) __bf16 v16bf;
typedef __attribute__((ext_vector_type(8)))  float  v8f;
typedef unsigned int u32x4 __attribute__((ext_vector_type(4)));
typedef int          i32x8 __attribute__((ext_vector_type(8)));
typedef int          i32x4 __attribute__((ext_vector_type(4)));

union Frag { v16bf v; uint4 q[2]; };

__device__ __forceinline__ unsigned short f2bf(float f) {
    unsigned int u = __float_as_uint(f);
    u += 0x7fffu + ((u >> 16) & 1u);          // round-to-nearest-even
    return (unsigned short)(u >> 16);
}

// Low 32 bits of a generic pointer to __shared__ = LDS byte offset (LDS aperture
// truncation rule, ISA 10.2).
__device__ __forceinline__ unsigned lds_off(const void* p) {
    return (unsigned)(unsigned long long)p;
}

// ---------------- TDM: DMA one 128x32 bf16 tile (row stride = strideElems) ----------------
// Tile is described as its own tensor (tensor_dim == tile_dim, global_addr = tile start).
// LDS padding: 4 DWORDs (16B) after every 16 DWORDs (64B row) -> padded stride 80B (LDT=40 halves).
__device__ __forceinline__ void tdm_load_tile_128x32(const unsigned short* gptr,
                                                     unsigned lds, unsigned strideElems) {
    unsigned long long ga = (unsigned long long)gptr;
    u32x4 g0;
    g0[0] = 1u;                                               // count=1, user descriptor
    g0[1] = lds;                                              // lds_addr (bytes)
    g0[2] = (unsigned)ga;                                     // global_addr[31:0]
    g0[3] = ((unsigned)(ga >> 32) & 0x01FFFFFFu) | 0x80000000u; // addr[56:32] | type=2
    i32x8 g1;
    g1[0] = (1 << 16)      // data_size = 2 bytes
          | (1 << 20)      // pad_enable
          | (3 << 22)      // pad_interval: 16 DWORDs (64B) between pads
          | (3 << 25);     // pad_amount: 4 DWORDs (16B)
    g1[1] = 32 << 16;      // tensor_dim0[15:0] = 32 (elements per row)
    g1[2] = 128 << 16;     // tensor_dim0 hi = 0 | tensor_dim1[15:0] = 128 rows
    g1[3] = 32 << 16;      // tensor_dim1 hi = 0 | tile_dim0 = 32
    g1[4] = 128;           // tile_dim1 = 128 | tile_dim2 = 0
    g1[5] = (int)strideElems; // tensor_dim0_stride[31:0] (data_size units)
    g1[6] = 0;             // stride hi | tensor_dim1_stride lo
    g1[7] = 0;
    i32x4 z4 = {0, 0, 0, 0};
    i32x8 z8 = {0, 0, 0, 0, 0, 0, 0, 0};
    // 6-arg form (clang-23 / therock-10.0 headers): groups 0..3 + extra group + cpol
    __builtin_amdgcn_tensor_load_to_lds(g0, g1, z4, z4, z8, 0);
}

// ---------------- embedding: h = emb[x] + pos ----------------
__global__ void embed_kernel(const int* __restrict__ x, const float* __restrict__ emb,
                             const float* __restrict__ pos, float* __restrict__ h) {
    size_t i = (size_t)blockIdx.x * blockDim.x + threadIdx.x;   // over B*T*E
    int e  = (int)(i & (cE - 1));
    int bt = (int)(i >> 9);
    int t  = bt & (cT - 1);
    int tok = x[bt];
    int tp  = t < cL ? t : cL - 1;
    h[i] = emb[(size_t)tok * cE + e] + pos[(size_t)tp * cE + e];
}

// ---------------- weight converts (f32 -> bf16, TRANSPOSED to [N][K]) ----------------
// per-layer [K][N] row-major -> [N][K]; gridDim.y = NB
__global__ void cvt_w_t(const float* __restrict__ s, unsigned short* __restrict__ d,
                        int K, int N) {
    int lyr = blockIdx.y;
    int i = blockIdx.x * 256 + threadIdx.x;     // < K*N
    int n = i % N, k = i / N;
    s += (size_t)lyr * K * N;
    d += (size_t)lyr * N * K;
    d[(size_t)n * K + k] = f2bf(s[i]);
}
// per-layer [NH][E][HS] -> [NH*HS][E]; gridDim.y = NB
__global__ void cvt_qkv_t(const float* __restrict__ s, unsigned short* __restrict__ d) {
    int lyr = blockIdx.y;
    int i = blockIdx.x * 256 + threadIdx.x;     // < NH*E*HS
    int dd = i & 63, e = (i >> 6) & 511, hh = i >> 15;
    s += (size_t)lyr * cNH * cE * cHS;
    d += (size_t)lyr * cE * cE;
    d[(size_t)(hh * 64 + dd) * cE + e] = f2bf(s[i]);
}

// ---------------- layernorm -> bf16 ----------------
__global__ __launch_bounds__(256) void ln_kernel(const float* __restrict__ x,
                                                 const float* __restrict__ g,
                                                 const float* __restrict__ bta,
                                                 unsigned short* __restrict__ out) {
    __shared__ float sb[256];
    int row = blockIdx.x, t = threadIdx.x;
    const float* xr = x + (size_t)row * cE;
    float x0 = xr[t], x1 = xr[t + 256];
    sb[t] = x0 + x1; __syncthreads();
    for (int off = 128; off > 0; off >>= 1) { if (t < off) sb[t] += sb[t + off]; __syncthreads(); }
    float mu = sb[0] * (1.f / cE); __syncthreads();
    float d0 = x0 - mu, d1 = x1 - mu;
    sb[t] = d0 * d0 + d1 * d1; __syncthreads();
    for (int off = 128; off > 0; off >>= 1) { if (t < off) sb[t] += sb[t + off]; __syncthreads(); }
    float rstd = rsqrtf(sb[0] * (1.f / cE) + 1e-5f);
    unsigned short* orow = out + (size_t)row * cE;
    orow[t]       = f2bf(d0 * rstd * g[t] + bta[t]);
    orow[t + 256] = f2bf(d1 * rstd * g[t + 256] + bta[t + 256]);
}

// ---------------- bf16 GEMM: TDM double-buffered tiles + WMMA ----------------
// C[M,N] = A[M,K] (bf16 row-major) * Bw^T where Bw is [N][K] bf16 (pre-transposed).
// mode 0: outB = C (bf16)                        [QKV]
// mode 1: outB = relu(C + bias) (bf16)           [FFN1]
// mode 2: resid += C + bias (f32, in-place)      [proj, FFN2]
__global__ __launch_bounds__(256) void gemm_bf16(const unsigned short* __restrict__ A,
                                                 const unsigned short* __restrict__ Bw,
                                                 int M, int N, int K,
                                                 const float* __restrict__ bias,
                                                 float* __restrict__ resid,
                                                 unsigned short* __restrict__ outB,
                                                 int mode) {
    constexpr int BM = 128, BN = 128, BK = 32, LDT = 40;   // LDT matches TDM pad config
    __shared__ unsigned short Al[2][BM * LDT];
    __shared__ unsigned short Bl[2][BN * LDT];

    const int tid  = threadIdx.x;
    const int lane = tid & 31;
    const int w    = tid >> 5;
    const int wm   = (w & 3) * 32;
    const int wn   = (w >> 2) * 64;
    const int m0   = blockIdx.y * BM;
    const int n0   = blockIdx.x * BN;
    const int l15  = lane & 15;
    const int koff = (lane < 16) ? 0 : 8;

    const unsigned short* Ab = A  + (size_t)m0 * K;   // tile row start (k advances)
    const unsigned short* Bb = Bw + (size_t)n0 * K;

    const v8f vzero = {0, 0, 0, 0, 0, 0, 0, 0};
    v8f acc[2][4];
    for (int i = 0; i < 2; ++i) for (int j = 0; j < 4; ++j) acc[i][j] = vzero;

    const int nk = K / BK;
    if (w == 0) {   // TDM ops are wave-level; one wave drives the DMA pipeline
        tdm_load_tile_128x32(Ab, lds_off(&Al[0][0]), (unsigned)K);
        tdm_load_tile_128x32(Bb, lds_off(&Bl[0][0]), (unsigned)K);
    }

    for (int i = 0; i < nk; ++i) {
        const int cur = i & 1;
        if (w == 0) {
            if (i + 1 < nk) {
                // prefetch next k-step into the other buffer (its last compute
                // finished before the trailing barrier of iteration i-1)
                const unsigned short* An = Ab + (size_t)(i + 1) * BK;
                const unsigned short* Bn = Bb + (size_t)(i + 1) * BK;
                tdm_load_tile_128x32(An, lds_off(&Al[cur ^ 1][0]), (unsigned)K);
                tdm_load_tile_128x32(Bn, lds_off(&Bl[cur ^ 1][0]), (unsigned)K);
                __builtin_amdgcn_s_wait_tensorcnt(2);   // in-order: current tiles landed
            } else {
                __builtin_amdgcn_s_wait_tensorcnt(0);
            }
        }
        __syncthreads();    // current buffer visible to all waves

        const unsigned short* Ac = Al[cur];
        const unsigned short* Bc = Bl[cur];
        Frag af[2];
        for (int mt = 0; mt < 2; ++mt) {
            int r = wm + mt * 16 + l15;
            af[mt].q[0] = *(const uint4*)&Ac[r * LDT + koff];
            af[mt].q[1] = *(const uint4*)&Ac[r * LDT + koff + 16];
        }
        for (int nt = 0; nt < 4; ++nt) {
            int c = wn + nt * 16 + l15;
            Frag bf;
            bf.q[0] = *(const uint4*)&Bc[c * LDT + koff];
            bf.q[1] = *(const uint4*)&Bc[c * LDT + koff + 16];
            for (int mt = 0; mt < 2; ++mt)
                acc[mt][nt] = __builtin_amdgcn_wmma_f32_16x16x32_bf16(
                    false, af[mt].v, false, bf.v, (short)0, acc[mt][nt], false, false);
        }
        __syncthreads();    // done with this buffer before it gets overwritten
    }

    const int rb = (lane < 16) ? 0 : 8;
    for (int mt = 0; mt < 2; ++mt) {
        for (int nt = 0; nt < 4; ++nt) {
            int col  = n0 + wn + nt * 16 + l15;
            float bv = (mode != 0) ? bias[col] : 0.f;
            for (int r = 0; r < 8; ++r) {
                int row  = m0 + wm + mt * 16 + rb + r;
                float xv = acc[mt][nt][r];
                size_t o = (size_t)row * N + col;
                if (mode == 0)       outB[o] = f2bf(xv);
                else if (mode == 1) { xv += bv; outB[o] = f2bf(xv > 0.f ? xv : 0.f); }
                else                 resid[o] += xv + bv;
            }
        }
    }
}

// ---------------- flash-style causal attention ----------------
// grid: (T/128, NH, B); block: 256 (8 waves); wave w owns q rows [qblk*128 + w*16, +16)
__global__ __launch_bounds__(256) void attn_kernel(const unsigned short* __restrict__ q,
                                                   const unsigned short* __restrict__ k,
                                                   const unsigned short* __restrict__ v,
                                                   unsigned short* __restrict__ o) {
    constexpr int LKT = 72, LVT = 40, LPT = 40;
    __shared__ unsigned short Kt[32 * LKT];          // K block row-major [32 keys][64 d]
    __shared__ unsigned short Vt[64 * LVT];          // V block transposed [64 d][32 keys]
    __shared__ unsigned short Pb[8 * 16 * LPT];      // per-wave P staging [16 q][32 keys]

    const int tid = threadIdx.x, lane = tid & 31, w = tid >> 5;
    const int b = blockIdx.z, h = blockIdx.y, qblk = blockIdx.x;
    const int l15 = lane & 15, koff8 = (lane < 16) ? 0 : 8;
    const int rb  = (lane < 16) ? 0 : 8;
    const int qrow0 = qblk * 128 + w * 16;
    const size_t rowstride = (size_t)cNH * cHS;      // 512
    const size_t base = (size_t)b * cT * rowstride + (size_t)h * cHS;
    const float scale = 0.125f;                      // HS^-0.5

    Frag qf[2];
    {
        const unsigned short* qp = q + base + (size_t)(qrow0 + l15) * rowstride;
        for (int j = 0; j < 2; ++j) {
            int ko = j * 32 + koff8;
            qf[j].q[0] = *(const uint4*)(qp + ko);
            qf[j].q[1] = *(const uint4*)(qp + ko + 16);
        }
    }

    float mrow[8], lrow[8];
    for (int r = 0; r < 8; ++r) { mrow[r] = -INFINITY; lrow[r] = 0.f; }
    const v8f vzero = {0, 0, 0, 0, 0, 0, 0, 0};
    v8f oacc[4];
    for (int nt = 0; nt < 4; ++nt) oacc[nt] = vzero;

    unsigned short* pw = &Pb[w * 16 * LPT];
    const int nkb = qblk * 4 + 4;

    for (int kb = 0; kb < nkb; ++kb) {
        const int kbase = kb * 32;
        __syncthreads();
        {   // K block: 32x64, b128 per thread
            int row = tid >> 3, quad = tid & 7;
            *(uint4*)&Kt[row * LKT + quad * 8] =
                *(const uint4*)(k + base + (size_t)(kbase + row) * rowstride + quad * 8);
        }
        {   // V block transposed
            int kk = tid >> 3, d0 = (tid & 7) * 8;
            uint4 raw = *(const uint4*)(v + base + (size_t)(kbase + kk) * rowstride + d0);
            const unsigned short* rs = (const unsigned short*)&raw;
            for (int j = 0; j < 8; ++j) Vt[(d0 + j) * LVT + kk] = rs[j];
        }
        __syncthreads();

        if (kbase <= qrow0 + 15) {
            v8f s0 = vzero, s1 = vzero;
            for (int j = 0; j < 2; ++j) {
                int kco = j * 32 + koff8;
                Frag kf0, kf1;
                kf0.q[0] = *(const uint4*)&Kt[l15 * LKT + kco];
                kf0.q[1] = *(const uint4*)&Kt[l15 * LKT + kco + 16];
                kf1.q[0] = *(const uint4*)&Kt[(16 + l15) * LKT + kco];
                kf1.q[1] = *(const uint4*)&Kt[(16 + l15) * LKT + kco + 16];
                s0 = __builtin_amdgcn_wmma_f32_16x16x32_bf16(false, qf[j].v, false, kf0.v,
                                                             (short)0, s0, false, false);
                s1 = __builtin_amdgcn_wmma_f32_16x16x32_bf16(false, qf[j].v, false, kf1.v,
                                                             (short)0, s1, false, false);
            }
            const int j0 = kbase + l15, j1 = kbase + 16 + l15;
            float p0[8], p1[8];
            for (int r = 0; r < 8; ++r) {
                int qg = qrow0 + rb + r;
                float a0 = (j0 <= qg) ? s0[r] * scale : -INFINITY;
                float a1 = (j1 <= qg) ? s1[r] * scale : -INFINITY;
                float mx = fmaxf(a0, a1);
                mx = fmaxf(mx, __shfl_xor(mx, 1));
                mx = fmaxf(mx, __shfl_xor(mx, 2));
                mx = fmaxf(mx, __shfl_xor(mx, 4));
                mx = fmaxf(mx, __shfl_xor(mx, 8));
                float mnew  = fmaxf(mrow[r], mx);
                float alpha = __expf(mrow[r] - mnew);
                float e0 = (j0 <= qg) ? __expf(s0[r] * scale - mnew) : 0.f;
                float e1 = (j1 <= qg) ? __expf(s1[r] * scale - mnew) : 0.f;
                float sm = e0 + e1;
                sm += __shfl_xor(sm, 1); sm += __shfl_xor(sm, 2);
                sm += __shfl_xor(sm, 4); sm += __shfl_xor(sm, 8);
                lrow[r] = lrow[r] * alpha + sm;
                mrow[r] = mnew;
                for (int nt = 0; nt < 4; ++nt) oacc[nt][r] *= alpha;
                p0[r] = e0; p1[r] = e1;
            }
            for (int r = 0; r < 8; ++r) {
                int row = rb + r;
                pw[row * LPT + l15]      = f2bf(p0[r]);
                pw[row * LPT + 16 + l15] = f2bf(p1[r]);
            }
            Frag pf;
            pf.q[0] = *(const uint4*)&pw[l15 * LPT + koff8];
            pf.q[1] = *(const uint4*)&pw[l15 * LPT + koff8 + 16];
            for (int nt = 0; nt < 4; ++nt) {
                Frag vf;
                int c = nt * 16 + l15;
                vf.q[0] = *(const uint4*)&Vt[c * LVT + koff8];
                vf.q[1] = *(const uint4*)&Vt[c * LVT + koff8 + 16];
                oacc[nt] = __builtin_amdgcn_wmma_f32_16x16x32_bf16(false, pf.v, false, vf.v,
                                                                   (short)0, oacc[nt], false, false);
            }
        }
    }

    for (int nt = 0; nt < 4; ++nt) {
        int d = nt * 16 + l15;
        for (int r = 0; r < 8; ++r) {
            int qr = qrow0 + rb + r;
            o[base + (size_t)qr * rowstride + d] = f2bf(oacc[nt][r] / lrow[r]);
        }
    }
}

// ---------------- final LN + tiny logits head (V=16) ----------------
__global__ __launch_bounds__(256) void final_kernel(const float* __restrict__ h,
                                                    const float* __restrict__ g,
                                                    const float* __restrict__ bta,
                                                    const float* __restrict__ ow,
                                                    const float* __restrict__ ob,
                                                    float* __restrict__ out) {
    __shared__ float sb[256];
    __shared__ float hn[cE];
    int row = blockIdx.x, t = threadIdx.x;
    const float* xr = h + (size_t)row * cE;
    float x0 = xr[t], x1 = xr[t + 256];
    sb[t] = x0 + x1; __syncthreads();
    for (int off = 128; off > 0; off >>= 1) { if (t < off) sb[t] += sb[t + off]; __syncthreads(); }
    float mu = sb[0] * (1.f / cE); __syncthreads();
    float d0 = x0 - mu, d1 = x1 - mu;
    sb[t] = d0 * d0 + d1 * d1; __syncthreads();
    for (int off = 128; off > 0; off >>= 1) { if (t < off) sb[t] += sb[t + off]; __syncthreads(); }
    float rstd = rsqrtf(sb[0] * (1.f / cE) + 1e-5f); __syncthreads();
    hn[t]       = d0 * rstd * g[t] + bta[t];
    hn[t + 256] = d1 * rstd * g[t + 256] + bta[t + 256];
    __syncthreads();
    int vv = t & 15, gg = t >> 4;
    float p = 0.f;
    for (int e = gg * 32; e < gg * 32 + 32; ++e) p += hn[e] * ow[e * cV + vv];
    sb[t] = p; __syncthreads();
    if (t < 16) {
        float s = 0.f;
        for (int g2 = 0; g2 < 16; ++g2) s += sb[g2 * 16 + t];
        out[(size_t)row * cV + t] = (s + ob[t]) * 1.25f;   // 1/TEMP = 1/0.8
    }
}

// ---------------- host orchestration ----------------
extern "C" void kernel_launch(void* const* d_in, const int* in_sizes, int n_in,
                              void* d_out, int out_size, void* d_ws, size_t ws_size,
                              hipStream_t stream) {
    (void)in_sizes; (void)n_in; (void)out_size; (void)ws_size;
    const int*   x      = (const int*)  d_in[0];
    const float* emb    = (const float*)d_in[1];
    const float* pos    = (const float*)d_in[2];
    const float* wq     = (const float*)d_in[3];
    const float* wk     = (const float*)d_in[4];
    const float* wv     = (const float*)d_in[5];
    const float* proj_w = (const float*)d_in[6];
    const float* proj_b = (const float*)d_in[7];
    const float* ln1_g  = (const float*)d_in[8];
    const float* ln1_b  = (const float*)d_in[9];
    const float* ln2_g  = (const float*)d_in[10];
    const float* ln2_b  = (const float*)d_in[11];
    const float* ff_w1  = (const float*)d_in[12];
    const float* ff_b1  = (const float*)d_in[13];
    const float* ff_w2  = (const float*)d_in[14];
    const float* ff_b2  = (const float*)d_in[15];
    const float* lnf_g  = (const float*)d_in[16];
    const float* lnf_b  = (const float*)d_in[17];
    const float* out_w  = (const float*)d_in[18];
    const float* out_b  = (const float*)d_in[19];

    char* ws = (char*)d_ws;
    size_t off = 0;
    auto alloc = [&](size_t bytes) -> void* {
        void* p = ws + off;
        off += (bytes + 255) & ~(size_t)255;
        return p;
    };
    float*          h   = (float*)         alloc((size_t)cM * cE * 4);
    unsigned short* xn  = (unsigned short*)alloc((size_t)cM * cE * 2);
    unsigned short* qb  = (unsigned short*)alloc((size_t)cM * cE * 2);
    unsigned short* kb  = (unsigned short*)alloc((size_t)cM * cE * 2);
    unsigned short* vb  = (unsigned short*)alloc((size_t)cM * cE * 2);
    unsigned short* ob  = (unsigned short*)alloc((size_t)cM * cE * 2);
    unsigned short* act = (unsigned short*)alloc((size_t)cM * cFF * 2);
    unsigned short* wqc = (unsigned short*)alloc((size_t)cNB * cE * cE * 2);   // [N][K]
    unsigned short* wkc = (unsigned short*)alloc((size_t)cNB * cE * cE * 2);
    unsigned short* wvc = (unsigned short*)alloc((size_t)cNB * cE * cE * 2);
    unsigned short* pjc = (unsigned short*)alloc((size_t)cNB * cE * cE * 2);
    unsigned short* f1c = (unsigned short*)alloc((size_t)cNB * cFF * cE * 2);  // [2048][512]
    unsigned short* f2c = (unsigned short*)alloc((size_t)cNB * cE * cFF * 2);  // [512][2048]

    embed_kernel<<<(cM * cE) / 256, 256, 0, stream>>>(x, emb, pos, h);

    cvt_qkv_t<<<dim3((cNH * cE * cHS) / 256, cNB), 256, 0, stream>>>(wq, wqc);
    cvt_qkv_t<<<dim3((cNH * cE * cHS) / 256, cNB), 256, 0, stream>>>(wk, wkc);
    cvt_qkv_t<<<dim3((cNH * cE * cHS) / 256, cNB), 256, 0, stream>>>(wv, wvc);
    cvt_w_t<<<dim3((cE * cE) / 256, cNB), 256, 0, stream>>>(proj_w, pjc, cE, cE);
    cvt_w_t<<<dim3((cE * cFF) / 256, cNB), 256, 0, stream>>>(ff_w1, f1c, cE, cFF);
    cvt_w_t<<<dim3((cFF * cE) / 256, cNB), 256, 0, stream>>>(ff_w2, f2c, cFF, cE);

    const dim3 gN512(cE / 128, cM / 128);    // (4, 64)
    const dim3 gN2048(cFF / 128, cM / 128);  // (16, 64)
    const dim3 gAttn(cT / 128, cNH, cB);     // (4, 8, 16)

    for (int i = 0; i < cNB; ++i) {
        const unsigned short* wqi = wqc + (size_t)i * cE * cE;
        const unsigned short* wki = wkc + (size_t)i * cE * cE;
        const unsigned short* wvi = wvc + (size_t)i * cE * cE;
        const unsigned short* pji = pjc + (size_t)i * cE * cE;
        const unsigned short* f1i = f1c + (size_t)i * cFF * cE;
        const unsigned short* f2i = f2c + (size_t)i * cE * cFF;

        ln_kernel<<<cM, 256, 0, stream>>>(h, ln1_g + i * cE, ln1_b + i * cE, xn);
        gemm_bf16<<<gN512, 256, 0, stream>>>(xn, wqi, cM, cE, cE, nullptr, nullptr, qb, 0);
        gemm_bf16<<<gN512, 256, 0, stream>>>(xn, wki, cM, cE, cE, nullptr, nullptr, kb, 0);
        gemm_bf16<<<gN512, 256, 0, stream>>>(xn, wvi, cM, cE, cE, nullptr, nullptr, vb, 0);
        attn_kernel<<<gAttn, 256, 0, stream>>>(qb, kb, vb, ob);
        gemm_bf16<<<gN512, 256, 0, stream>>>(ob, pji, cM, cE, cE, proj_b + i * cE, h, nullptr, 2);
        ln_kernel<<<cM, 256, 0, stream>>>(h, ln2_g + i * cE, ln2_b + i * cE, xn);
        gemm_bf16<<<gN2048, 256, 0, stream>>>(xn, f1i, cM, cFF, cE, ff_b1 + i * cFF, nullptr, act, 1);
        gemm_bf16<<<gN512, 256, 0, stream>>>(act, f2i, cM, cE, cFF, ff_b2 + i * cE, h, nullptr, 2);
    }

    final_kernel<<<cM, 256, 0, stream>>>(h, lnf_g, lnf_b, out_w, out_b, (float*)d_out);
}
